// SelfNeighborAttention_4294967296540
// MI455X (gfx1250) — matrix operand order repaired and checked by reading, not compiled, not run on hardware
//
#include <hip/hip_runtime.h>
#include <hip/hip_bf16.h>
#include <cstdint>

#define B_CELLS 4096
#define KNB     32
#define DDIM    1024
#define RDIM    128
#define ALPHA_C 2.0f
#define RADIUS_C 100.0f
#define THRESH_C 0.01f

// ---------------------------------------------------------------------------
// CDNA5 types / feature detection
// ---------------------------------------------------------------------------
typedef __bf16 v16bf __attribute__((ext_vector_type(16)));
typedef float  v8f   __attribute__((ext_vector_type(8)));

// Async global->LDS staging via inline CDNA5 asm (ISA 15.18.3 opcode 98).
// The clang builtin has toolchain-dependent signatures, so we bypass it.
#if defined(__gfx1250__)
#define USE_ASYNC_LDS_ASM 1
#endif

__device__ __forceinline__ unsigned short f2bf(float f) {
  unsigned int u = __float_as_uint(f);
  u = u + 0x7FFFu + ((u >> 16) & 1u);   // round-to-nearest-even
  return (unsigned short)(u >> 16);
}
__device__ __forceinline__ float bf2f(unsigned short h) {
  return __uint_as_float(((unsigned int)h) << 16);
}

union Frag {
  uint4 u[2];
  v16bf v;
};

__device__ __forceinline__ void wait_async_lds() {
#ifdef USE_ASYNC_LDS_ASM
  asm volatile("s_wait_asynccnt 0" ::: "memory");
#endif
}

// Stage a 128-row x 32-col bf16 tile (64B per row) global -> LDS.
// 512 16B chunks, 256 threads x 2.
__device__ __forceinline__ void stage_tile(const unsigned short* __restrict__ g,
                                           int rowStride, unsigned short* l, int tid) {
#pragma unroll
  for (int i = 0; i < 2; ++i) {
    const int c   = tid + 256 * i;
    const int row = c >> 2;
    const int seg = c & 3;
    const unsigned short* gp = g + (size_t)row * rowStride + seg * 8;
    unsigned short*       lp = l + row * 32 + seg * 8;
#ifdef USE_ASYNC_LDS_ASM
    const unsigned           lds_off = (unsigned)(uintptr_t)lp;
    const unsigned long long gaddr   = (unsigned long long)(uintptr_t)gp;
    asm volatile("global_load_async_to_lds_b128 %0, %1, off"
                 :
                 : "v"(lds_off), "v"(gaddr)
                 : "memory");
#else
    *(uint4*)lp = *(const uint4*)gp;
#endif
  }
}

// ---------------------------------------------------------------------------
// Kernel 1: fused LayerNorm + bf16 cast (target rows + neighbor rows)
// ---------------------------------------------------------------------------
__global__ __launch_bounds__(256) void ln_cast_kernel(
    const float* __restrict__ target, const float* __restrict__ neighbors,
    const float* __restrict__ ln_scale, const float* __restrict__ ln_bias,
    float* __restrict__ tc_f32, unsigned short* __restrict__ tc_bf16,
    unsigned short* __restrict__ nb_bf16) {
  const int row = blockIdx.x;
  const int tid = threadIdx.x;
  const bool isTarget = row < B_CELLS;
  const float* src = isTarget ? target + (size_t)row * DDIM
                              : neighbors + (size_t)(row - B_CELLS) * DDIM;
  float4 x = ((const float4*)src)[tid];
  float s  = x.x + x.y + x.z + x.w;
  float s2 = x.x * x.x + x.y * x.y + x.z * x.z + x.w * x.w;
#pragma unroll
  for (int off = 16; off > 0; off >>= 1) {
    s  += __shfl_down(s, off);
    s2 += __shfl_down(s2, off);
  }
  __shared__ float rs[8], rq[8];
  if ((tid & 31) == 0) { rs[tid >> 5] = s; rq[tid >> 5] = s2; }
  __syncthreads();
  s = 0.f; s2 = 0.f;
#pragma unroll
  for (int i = 0; i < 8; ++i) { s += rs[i]; s2 += rq[i]; }
  const float mu   = s * (1.0f / DDIM);
  const float var  = fmaxf(s2 * (1.0f / DDIM) - mu * mu, 0.0f);
  const float rinv = rsqrtf(var + 1e-5f);
  float4 sc = ((const float4*)ln_scale)[tid];
  float4 bi = ((const float4*)ln_bias)[tid];
  float4 y;
  y.x = (x.x - mu) * rinv * sc.x + bi.x;
  y.y = (x.y - mu) * rinv * sc.y + bi.y;
  y.z = (x.z - mu) * rinv * sc.z + bi.z;
  y.w = (x.w - mu) * rinv * sc.w + bi.w;
  uint2 p;
  p.x = (unsigned)f2bf(y.x) | ((unsigned)f2bf(y.y) << 16);
  p.y = (unsigned)f2bf(y.z) | ((unsigned)f2bf(y.w) << 16);
  if (isTarget) {
    ((float4*)(tc_f32 + (size_t)row * DDIM))[tid] = y;
    ((uint2*)(tc_bf16 + (size_t)row * DDIM))[tid] = p;
  } else {
    ((uint2*)(nb_bf16 + (size_t)(row - B_CELLS) * DDIM))[tid] = p;
  }
}

// ---------------------------------------------------------------------------
// Kernel 2: fp32 -> bf16 weight conversion
// ---------------------------------------------------------------------------
__global__ __launch_bounds__(256) void cvt_f32_bf16_kernel(
    const float* __restrict__ in, unsigned short* __restrict__ out, int n) {
  const int i = blockIdx.x * 256 + threadIdx.x;
  if (i < n) out[i] = f2bf(in[i]);
}

// ---------------------------------------------------------------------------
// Kernel 3: bf16 WMMA GEMM   out[m,n] = sum_k X[m,k]*W[n,k] + bias[n]
// block tile 128x128, 8 waves (4M x 2N), wave tile 32x64, k-step 32,
// double-buffered async LDS staging.
// EPI: 0 = bias only, 1 = bias + leaky_relu(0.01)
// ---------------------------------------------------------------------------
template <int EPI>
__global__ __launch_bounds__(256) void gemm_bf16_wmma(
    const unsigned short* __restrict__ X, const unsigned short* __restrict__ W,
    const float* __restrict__ bias, float* __restrict__ out,
    int Ntot, int Kdim) {
  __shared__ __align__(16) unsigned short lA[2][128 * 32];
  __shared__ __align__(16) unsigned short lB[2][128 * 32];

  const int tid  = threadIdx.x;
  const int lane = tid & 31;
  const int wave = tid >> 5;
  const int wm   = wave >> 1;   // 0..3  -> M offset wm*32
  const int wn   = wave & 1;    // 0..1  -> N offset wn*64
  const int half = lane >> 4;   // lane group 0/1
  const int r    = lane & 15;

  const size_t m0 = (size_t)blockIdx.y * 128;
  const int    n0 = blockIdx.x * 128;

  const unsigned short* Xb = X + m0 * Kdim;
  const unsigned short* Wb = W + (size_t)n0 * Kdim;

  const v8f vzero = {0.f, 0.f, 0.f, 0.f, 0.f, 0.f, 0.f, 0.f};
  v8f acc[2][4];
#pragma unroll
  for (int mt = 0; mt < 2; ++mt)
#pragma unroll
    for (int nt = 0; nt < 4; ++nt)
      acc[mt][nt] = vzero;

  const int ksteps = Kdim >> 5;
  stage_tile(Xb, Kdim, &lA[0][0], tid);
  stage_tile(Wb, Kdim, &lB[0][0], tid);

  for (int ks = 0; ks < ksteps; ++ks) {
    const int cur = ks & 1;
    wait_async_lds();
    __syncthreads();
    if (ks + 1 < ksteps) {
      stage_tile(Xb + (ks + 1) * 32, Kdim, &lA[cur ^ 1][0], tid);
      stage_tile(Wb + (ks + 1) * 32, Kdim, &lB[cur ^ 1][0], tid);
    }

    // A fragments (16x32 bf16): lanes 0-15 row=r, k 0..7 & 16..23;
    //                           lanes 16-31 row=r, k 8..15 & 24..31.
    Frag a[2];
#pragma unroll
    for (int mt = 0; mt < 2; ++mt) {
      const unsigned short* p = &lA[cur][0] + (wm * 32 + mt * 16 + r) * 32 + half * 8;
      a[mt].u[0] = *(const uint4*)(p);
      a[mt].u[1] = *(const uint4*)(p + 16);
    }
    // B fragments (32x16 bf16): N = lane%16; lanes 0-15 k 0..7 & 16..23, lanes 16-31 k 8..15 & 24..31.
    Frag b[4];
#pragma unroll
    for (int nt = 0; nt < 4; ++nt) {
      const unsigned short* p = &lB[cur][0] + (wn * 64 + nt * 16 + r) * 32 + half * 8;
      b[nt].u[0] = *(const uint4*)(p);
      b[nt].u[1] = *(const uint4*)(p + 16);
    }
#pragma unroll
    for (int mt = 0; mt < 2; ++mt)
#pragma unroll
      for (int nt = 0; nt < 4; ++nt)
        acc[mt][nt] = __builtin_amdgcn_wmma_f32_16x16x32_bf16(
            false, a[mt].v, false, b[nt].v, (short)0, acc[mt][nt], false, false);
  }

  // Epilogue: C/D layout -> col = base + (lane&15); VGPR i -> row = base + half*8 + i
#pragma unroll
  for (int mt = 0; mt < 2; ++mt)
#pragma unroll
    for (int nt = 0; nt < 4; ++nt) {
      const int col = n0 + wn * 64 + nt * 16 + r;
      const float bv = bias[col];
#pragma unroll
      for (int i = 0; i < 8; ++i) {
        const size_t rowm = m0 + wm * 32 + mt * 16 + half * 8 + i;
        float v = acc[mt][nt][i] + bv;
        if (EPI == 1) v = v > 0.f ? v : 0.01f * v;
        out[rowm * (size_t)Ntot + col] = v;
      }
    }
}

// ---------------------------------------------------------------------------
// Kernel 4: distance weighting + 33-way softmax + threshold + context (fused)
// in-place on self_w / nb_w regions of d_out; writes bf16 context.
// ---------------------------------------------------------------------------
__global__ __launch_bounds__(256) void softmax_ctx_kernel(
    const float* __restrict__ distances, const float* __restrict__ beta_p,
    const float* __restrict__ tc_f32, const unsigned short* __restrict__ nb_bf16,
    float* __restrict__ self_w /* in: self_scores */,
    float* __restrict__ nb_w   /* in: nb_scores  */,
    unsigned short* __restrict__ ctx_bf16) {
  const int g = blockIdx.x * 256 + threadIdx.x;   // < B*D
  const int b = g >> 10;
  const int d = g & (DDIM - 1);

  __shared__ float sdw[KNB];
  if (threadIdx.x < KNB)
    sdw[threadIdx.x] =
        __expf(-ALPHA_C * distances[(size_t)b * KNB + threadIdx.x] / (RADIUS_C + 1e-8f));
  __syncthreads();

  const float s = self_w[g];
  float v[KNB];
  float mx = s;
#pragma unroll
  for (int k = 0; k < KNB; ++k) {
    const size_t idx = (size_t)(b * KNB + k) * DDIM + d;
    v[k] = nb_w[idx] * sdw[k];
    mx = fmaxf(mx, v[k]);
  }
  float e0 = __expf(s - mx);
  float sum = e0;
#pragma unroll
  for (int k = 0; k < KNB; ++k) {
    v[k] = __expf(v[k] - mx);
    sum += v[k];
  }
  const float inv  = 1.0f / sum;
  const float beta = beta_p[0];
  float w0 = e0 * inv;
  if (w0 < THRESH_C) w0 = 0.f;
  self_w[g] = w0;
  float ctx = beta * w0 * tc_f32[g];
  const float ob = 1.0f - beta;
#pragma unroll
  for (int k = 0; k < KNB; ++k) {
    const size_t idx = (size_t)(b * KNB + k) * DDIM + d;
    float w = v[k] * inv;
    if (w < THRESH_C) w = 0.f;
    nb_w[idx] = w;
    ctx += ob * w * bf2f(nb_bf16[idx]);
  }
  ctx_bf16[g] = f2bf(ctx);
}

// ---------------------------------------------------------------------------
// host launcher
// ---------------------------------------------------------------------------
extern "C" void kernel_launch(void* const* d_in, const int* in_sizes, int n_in,
                              void* d_out, int out_size, void* d_ws, size_t ws_size,
                              hipStream_t stream) {
  (void)in_sizes; (void)n_in; (void)out_size; (void)ws_size;
  const float* target    = (const float*)d_in[0];
  const float* neighbors = (const float*)d_in[1];
  const float* distances = (const float*)d_in[2];
  const float* beta      = (const float*)d_in[3];
  const float* ln_scale  = (const float*)d_in[4];
  const float* ln_bias   = (const float*)d_in[5];
  const float* w_self    = (const float*)d_in[6];
  const float* b_self    = (const float*)d_in[7];
  const float* w_nb      = (const float*)d_in[8];
  const float* b_nb      = (const float*)d_in[9];
  const float* w_red     = (const float*)d_in[10];
  const float* b_red     = (const float*)d_in[11];

  float* out_red   = (float*)d_out;                               // B*R
  float* out_selfw = out_red + (size_t)B_CELLS * RDIM;            // B*D
  float* out_nbw   = out_selfw + (size_t)B_CELLS * DDIM;          // B*K*D

  const size_t BD  = (size_t)B_CELLS * DDIM;
  const size_t BKD = (size_t)B_CELLS * KNB * DDIM;
  const size_t DD  = (size_t)DDIM * DDIM;
  const size_t RD  = (size_t)RDIM * DDIM;

  char* ws = (char*)d_ws;
  float*          tc_f32   = (float*)ws;          ws += BD * sizeof(float);
  unsigned short* tc_bf    = (unsigned short*)ws; ws += BD * 2;
  unsigned short* nb_bf    = (unsigned short*)ws; ws += BKD * 2;
  unsigned short* wself_bf = (unsigned short*)ws; ws += DD * 2;
  unsigned short* wnb_bf   = (unsigned short*)ws; ws += DD * 2;
  unsigned short* wred_bf  = (unsigned short*)ws; ws += RD * 2;
  unsigned short* ctx_bf   = (unsigned short*)ws; ws += BD * 2;

  // 1) LayerNorm + bf16 cast
  ln_cast_kernel<<<B_CELLS * (KNB + 1), 256, 0, stream>>>(
      target, neighbors, ln_scale, ln_bias, tc_f32, tc_bf, nb_bf);

  // 2) weight conversion
  cvt_f32_bf16_kernel<<<(int)((DD + 255) / 256), 256, 0, stream>>>(w_self, wself_bf, (int)DD);
  cvt_f32_bf16_kernel<<<(int)((DD + 255) / 256), 256, 0, stream>>>(w_nb,   wnb_bf,   (int)DD);
  cvt_f32_bf16_kernel<<<(int)((RD + 255) / 256), 256, 0, stream>>>(w_red,  wred_bf,  (int)RD);

  // 3) self scores -> self_w region
  {
    dim3 g(DDIM / 128, B_CELLS / 128);
    gemm_bf16_wmma<0><<<g, 256, 0, stream>>>(tc_bf, wself_bf, b_self, out_selfw, DDIM, DDIM);
  }
  // 4) neighbor scores -> nb_w region
  {
    dim3 g(DDIM / 128, (B_CELLS * KNB) / 128);
    gemm_bf16_wmma<0><<<g, 256, 0, stream>>>(nb_bf, wnb_bf, b_nb, out_nbw, DDIM, DDIM);
  }
  // 5) softmax + threshold + context
  softmax_ctx_kernel<<<(int)(BD / 256), 256, 0, stream>>>(
      distances, beta, tc_f32, nb_bf, out_selfw, out_nbw, ctx_bf);

  // 6) reduction GEMM + leaky relu -> reduced
  {
    dim3 g(RDIM / 128, B_CELLS / 128);
    gemm_bf16_wmma<1><<<g, 256, 0, stream>>>(ctx_bf, wred_bf, b_red, out_red, RDIM, DDIM);
  }
}
